// SpikeAttention_82386062672604
// MI455X (gfx1250) — matrix operand order repaired and checked by reading, not compiled, not run on hardware
//
#include <hip/hip_runtime.h>
#include <math.h>

// ---------------------------------------------------------------------------
// CDNA5 (gfx1250) wave32 WMMA fragments, v_wmma_f32_16x16x32_f16
// ---------------------------------------------------------------------------
typedef __attribute__((ext_vector_type(16))) _Float16 v16h;
typedef __attribute__((ext_vector_type(8)))  float    v8f;

#define DEV static __device__ __forceinline__

DEV v8f wmma16(v16h a, v16h b, v8f c) {
  return __builtin_amdgcn_wmma_f32_16x16x32_f16(false, a, false, b, (short)0, c,
                                                false, false);
}

// A fragment (16x32 f16): row-major storage s[m][k], leading dim lda (halves).
// lane<16: m=lane, K {0..15}; lane>=16: m=lane-16, K {16..31}.
DEV v16h frag_a(const _Float16* s, int lda, int m0, int k0, int lane) {
  int m  = m0 + (lane & 15);
  int kb = k0 + ((lane >> 4) << 4);
  const _Float16* p = s + m * lda + kb;
  v16h r;
#pragma unroll
  for (int j = 0; j < 8; ++j) { r[j] = p[j]; r[8 + j] = p[8 + j]; }
  return r;
}

// A fragment when the matrix is stored transposed: s[k][m] (A[m][k] = s[k][m]).
DEV v16h frag_aT(const _Float16* s, int lda, int m0, int k0, int lane) {
  int m  = m0 + (lane & 15);
  int kb = k0 + ((lane >> 4) << 4);
  v16h r;
#pragma unroll
  for (int j = 0; j < 8; ++j) {
    r[j]     = s[(kb + j) * lda + m];
    r[8 + j] = s[(kb + 8 + j) * lda + m];
  }
  return r;
}

// B fragment (32x16 f16): row-major K x N storage s[k][n].
// VGPR v holds K=2v,2v+1 (lanes 0-15) / K=16+2v,17+2v (lanes 16-31).
DEV v16h frag_b(const _Float16* s, int ldb, int k0, int n0, int lane) {
  int n  = n0 + (lane & 15);
  int ko = k0 + ((lane >> 4) << 4);
  v16h r;
#pragma unroll
  for (int v = 0; v < 8; ++v) {
    r[2 * v]     = s[(ko + 2 * v) * ldb + n];
    r[2 * v + 1] = s[(ko + 2 * v + 1) * ldb + n];
  }
  return r;
}

// B fragment from N-major storage: s[n][k] == B[k][n] (i.e. B^T row-major).
DEV v16h frag_bt(const _Float16* s, int ldb, int k0, int n0, int lane) {
  int n  = n0 + (lane & 15);
  int ko = k0 + ((lane >> 4) << 4);
  const _Float16* p = s + n * ldb + ko;
  v16h r;
#pragma unroll
  for (int j = 0; j < 16; ++j) r[j] = p[j];
  return r;
}

// ---------------------------------------------------------------------------
// Problem constants
// ---------------------------------------------------------------------------
#define N_TOK 1024
#define DMODEL 1024
#define NHEAD 8
#define HD 128
#define FD 256
#define SD 512
#define CHUNK 128
#define NCHUNK 8

// ---------------------------------------------------------------------------
// Spike-path scalar kernels (tiny FLOPs)
// ---------------------------------------------------------------------------
__global__ void spike_ln_kernel(const float* __restrict__ spike,
                                const float* __restrict__ g,
                                const float* __restrict__ b,
                                float* __restrict__ out) {
  __shared__ float red[SD];
  int t = threadIdx.x;
  float v = spike[t];
  red[t] = v; __syncthreads();
  for (int s = SD / 2; s > 0; s >>= 1) { if (t < s) red[t] += red[t + s]; __syncthreads(); }
  float m = red[0] / (float)SD;
  __syncthreads();
  float d = v - m;
  red[t] = d * d; __syncthreads();
  for (int s = SD / 2; s > 0; s >>= 1) { if (t < s) red[t] += red[t + s]; __syncthreads(); }
  float var = red[0] / (float)SD;
  out[t] = d * rsqrtf(var + 1e-5f) * g[t] + b[t];
}

// out[d] = (act? gelu : id)(dot(vin[0..K), W[d][0..K)) + bias[d]); grid*256 == D
__global__ __launch_bounds__(256) void matvec_kernel(const float* __restrict__ vin,
                                                     const float* __restrict__ W,
                                                     const float* __restrict__ bias,
                                                     float* __restrict__ out,
                                                     int K, int act) {
  __shared__ float sv[DMODEL];
  for (int i = threadIdx.x; i < K; i += 256) sv[i] = vin[i];
  __syncthreads();
  int d = blockIdx.x * 256 + threadIdx.x;
  const float* wr = W + (size_t)d * K;
  float s = 0.f;
  for (int k = 0; k < K; ++k) s += sv[k] * wr[k];
  s += bias[d];
  if (act) s = 0.5f * s * (1.f + erff(s * 0.70710678118654752f));
  out[d] = s;
}

// LN(h2; g_ln,b_ln) -> sD; then stats of sD give p[] and per-row alpha[].
__global__ void stats_kernel(const float* __restrict__ h2,
                             const float* __restrict__ g_ln,
                             const float* __restrict__ b_ln,
                             const float* __restrict__ g_sn2,
                             const float* __restrict__ sm,
                             float* __restrict__ p, float* __restrict__ alpha) {
  __shared__ float red[DMODEL];
  int t = threadIdx.x;
  float v = h2[t];
  red[t] = v; __syncthreads();
  for (int s = DMODEL / 2; s > 0; s >>= 1) { if (t < s) red[t] += red[t + s]; __syncthreads(); }
  float m = red[0] / (float)DMODEL;
  __syncthreads();
  float d = v - m;
  red[t] = d * d; __syncthreads();
  for (int s = DMODEL / 2; s > 0; s >>= 1) { if (t < s) red[t] += red[t + s]; __syncthreads(); }
  float var = red[0] / (float)DMODEL;
  float sD = d * rsqrtf(var + 1e-5f) * g_ln[t] + b_ln[t];
  __syncthreads();
  red[t] = sD; __syncthreads();
  for (int s = DMODEL / 2; s > 0; s >>= 1) { if (t < s) red[t] += red[t + s]; __syncthreads(); }
  float m2 = red[0] / (float)DMODEL;
  __syncthreads();
  float d2 = sD - m2;
  red[t] = d2 * d2; __syncthreads();
  for (int s = DMODEL / 2; s > 0; s >>= 1) { if (t < s) red[t] += red[t + s]; __syncthreads(); }
  float v2 = red[0] / (float)DMODEL;
  p[t] = d2 * g_sn2[t];
  float c = sm[t];
  alpha[t] = c * rsqrtf(c * c * v2 + 1e-5f);
}

// kpv = p @ W2^T ; kcv = b_sn2 @ W2^T + b2     (the whole K GEMM, rank-collapsed)
__global__ __launch_bounds__(256) void kvec_kernel(const float* __restrict__ p,
                                                   const float* __restrict__ bsn2,
                                                   const float* __restrict__ W2,
                                                   const float* __restrict__ b2,
                                                   float* __restrict__ kpv,
                                                   float* __restrict__ kcv) {
  __shared__ float sp[DMODEL], sb[DMODEL];
  for (int i = threadIdx.x; i < DMODEL; i += 256) { sp[i] = p[i]; sb[i] = bsn2[i]; }
  __syncthreads();
  int d = blockIdx.x * 256 + threadIdx.x;
  const float* wr = W2 + (size_t)d * DMODEL;
  float s1 = 0.f, s2 = 0.f;
  for (int k = 0; k < DMODEL; ++k) { float w = wr[k]; s1 += sp[k] * w; s2 += sb[k] * w; }
  kpv[d] = s1;
  kcv[d] = s2 + b2[d];
}

// Af[h][f] = proj_h^T kpv_h ; Bf[h][f] = proj_h^T kcv_h
__global__ __launch_bounds__(256) void featk_kernel(const float* __restrict__ kpv,
                                                    const float* __restrict__ kcv,
                                                    const float* __restrict__ proj,
                                                    float* __restrict__ Af,
                                                    float* __restrict__ Bf) {
  __shared__ float skp[HD], skc[HD];
  int h = blockIdx.x, f = threadIdx.x;
  if (f < HD) { skp[f] = kpv[h * HD + f]; skc[f] = kcv[h * HD + f]; }
  __syncthreads();
  const float* pr = proj + (size_t)h * HD * FD;
  float a = 0.f, b = 0.f;
  for (int hd = 0; hd < HD; ++hd) { float w = pr[hd * FD + f]; a += skp[hd] * w; b += skc[hd] * w; }
  Af[h * FD + f] = a;
  Bf[h * FD + f] = b;
}

// kp[h][i][f] = elu(alpha_i*Af + Bf) + 1
__global__ __launch_bounds__(256) void kpmat_kernel(const float* __restrict__ alpha,
                                                    const float* __restrict__ Af,
                                                    const float* __restrict__ Bf,
                                                    float* __restrict__ kp) {
  int idx = blockIdx.x;                // h*1024 + i
  int h = idx >> 10, i = idx & 1023;
  int f = threadIdx.x;
  float x = alpha[i] * Af[h * FD + f] + Bf[h * FD + f];
  kp[(size_t)idx * FD + f] = (x > 0.f) ? (x + 1.f) : __expf(x);
}

// ---------------------------------------------------------------------------
// Dense GEMM: OUT = X @ W^T + bias   (M=N=K=1024), WMMA f16->f32
// mode 0: out[m][n] (N,D);  mode 1: out[h][m][hd] heads layout
// ---------------------------------------------------------------------------
__global__ __launch_bounds__(256) void gemm_xwT_kernel(const float* __restrict__ X,
                                                       const float* __restrict__ W,
                                                       const float* __restrict__ bias,
                                                       float* __restrict__ out,
                                                       int K, int Nn, int mode) {
  __shared__ _Float16 sa[128 * 34];
  __shared__ _Float16 sbt[128 * 34];
  const int tid = threadIdx.x, lane = tid & 31, wave = tid >> 5;
  const int wm = wave >> 1, wn = wave & 1;
  const int m0g = blockIdx.y * 128, n0g = blockIdx.x * 128;
  const int lrow = tid >> 1, lcol = (tid & 1) * 16;
  v8f acc[2][4];
#pragma unroll
  for (int i = 0; i < 2; ++i)
#pragma unroll
    for (int j = 0; j < 4; ++j) acc[i][j] = (v8f){0, 0, 0, 0, 0, 0, 0, 0};

  for (int k0 = 0; k0 < K; k0 += 32) {
    const float* ap = X + (size_t)(m0g + lrow) * K + k0 + lcol;
    const float* bp = W + (size_t)(n0g + lrow) * K + k0 + lcol;
    _Float16* da = sa + lrow * 34 + lcol;
    _Float16* db = sbt + lrow * 34 + lcol;
#pragma unroll
    for (int j = 0; j < 16; ++j) { da[j] = (_Float16)ap[j]; db[j] = (_Float16)bp[j]; }
    __syncthreads();
    v16h a0 = frag_a(sa, 34, wm * 32, 0, lane);
    v16h a1 = frag_a(sa, 34, wm * 32 + 16, 0, lane);
#pragma unroll
    for (int j = 0; j < 4; ++j) {
      v16h b = frag_bt(sbt, 34, 0, wn * 64 + j * 16, lane);
      acc[0][j] = wmma16(a0, b, acc[0][j]);
      acc[1][j] = wmma16(a1, b, acc[1][j]);
    }
    __syncthreads();
  }
  const int msel = (lane >> 4) << 3, nsel = lane & 15;
#pragma unroll
  for (int i = 0; i < 2; ++i)
#pragma unroll
    for (int j = 0; j < 4; ++j)
#pragma unroll
      for (int r = 0; r < 8; ++r) {
        int m = m0g + wm * 32 + i * 16 + msel + r;
        int n = n0g + wn * 64 + j * 16 + nsel;
        float v = acc[i][j][r] + bias[n];
        if (mode == 0) out[(size_t)m * Nn + n] = v;
        else {
          int h = n >> 7;
          out[(((size_t)h << 10) + m) * HD + (n & 127)] = v;
        }
      }
}

// ---------------------------------------------------------------------------
// qp = elu(Qh @ proj) + 1 : per head (1024x128)@(128x256), WMMA
// ---------------------------------------------------------------------------
__global__ __launch_bounds__(256) void featq_kernel(const float* __restrict__ Qh,
                                                    const float* __restrict__ proj,
                                                    float* __restrict__ qp) {
  __shared__ _Float16 sa[128 * 34];
  __shared__ _Float16 sb[32 * 130];
  const int h = blockIdx.z, m0 = blockIdx.y * 128, n0 = blockIdx.x * 128;
  const int tid = threadIdx.x, lane = tid & 31, wave = tid >> 5;
  const int wm = wave >> 1, wn = wave & 1;
  const float* A = Qh + (size_t)h * N_TOK * HD;
  const float* B = proj + (size_t)h * HD * FD;
  v8f acc[2][4];
#pragma unroll
  for (int i = 0; i < 2; ++i)
#pragma unroll
    for (int j = 0; j < 4; ++j) acc[i][j] = (v8f){0, 0, 0, 0, 0, 0, 0, 0};

  const int lrow = tid >> 1, lcol = (tid & 1) * 16;
  const int brow = tid >> 3, bcol = (tid & 7) * 16;
  for (int kk = 0; kk < 4; ++kk) {
    int k0 = kk * 32;
    const float* ap = A + (size_t)(m0 + lrow) * HD + k0 + lcol;
    _Float16* da = sa + lrow * 34 + lcol;
#pragma unroll
    for (int j = 0; j < 16; ++j) da[j] = (_Float16)ap[j];
    const float* bp = B + (size_t)(k0 + brow) * FD + n0 + bcol;
    _Float16* db = sb + brow * 130 + bcol;
#pragma unroll
    for (int j = 0; j < 16; ++j) db[j] = (_Float16)bp[j];
    __syncthreads();
    v16h a0 = frag_a(sa, 34, wm * 32, 0, lane);
    v16h a1 = frag_a(sa, 34, wm * 32 + 16, 0, lane);
#pragma unroll
    for (int j = 0; j < 4; ++j) {
      v16h b = frag_b(sb, 130, 0, wn * 64 + j * 16, lane);
      acc[0][j] = wmma16(a0, b, acc[0][j]);
      acc[1][j] = wmma16(a1, b, acc[1][j]);
    }
    __syncthreads();
  }
  const int msel = (lane >> 4) << 3, nsel = lane & 15;
#pragma unroll
  for (int i = 0; i < 2; ++i)
#pragma unroll
    for (int j = 0; j < 4; ++j)
#pragma unroll
      for (int r = 0; r < 8; ++r) {
        int m = m0 + wm * 32 + i * 16 + msel + r;
        int n = n0 + wn * 64 + j * 16 + nsel;
        float x = acc[i][j][r];
        qp[(((size_t)h << 10) + m) * FD + n] = (x > 0.f) ? (x + 1.f) : __expf(x);
      }
}

// ---------------------------------------------------------------------------
// Pass A: per (head, chunk): S_c = KP_c^T @ V_c (256x128), z_c = colsum(KP_c)
// ---------------------------------------------------------------------------
__global__ __launch_bounds__(256) void chunk_kv_kernel(const float* __restrict__ kp,
                                                       const float* __restrict__ Vh,
                                                       float* __restrict__ Sc,
                                                       float* __restrict__ zc) {
  __shared__ _Float16 skp[32 * 258];  // [token][f]
  __shared__ _Float16 sv[32 * 130];   // [token][hd]
  const int h = blockIdx.x >> 3, c = blockIdx.x & 7;
  const int tid = threadIdx.x, lane = tid & 31, wave = tid >> 5;
  const int wm = wave >> 1, wn = wave & 1;  // wave tile 64x64
  const float* kpb = kp + (((size_t)h << 10) + c * CHUNK) * FD;
  const float* vb = Vh + (((size_t)h << 10) + c * CHUNK) * HD;
  v8f acc[4][4];
#pragma unroll
  for (int i = 0; i < 4; ++i)
#pragma unroll
    for (int j = 0; j < 4; ++j) acc[i][j] = (v8f){0, 0, 0, 0, 0, 0, 0, 0};

  for (int kk = 0; kk < 4; ++kk) {
    int t0 = kk * 32;
    { int r = tid >> 3, c0 = (tid & 7) * 32;
      const float* p = kpb + (size_t)(t0 + r) * FD + c0;
      _Float16* d = skp + r * 258 + c0;
#pragma unroll
      for (int j = 0; j < 32; ++j) d[j] = (_Float16)p[j]; }
    { int r = tid >> 3, c0 = (tid & 7) * 16;
      const float* p = vb + (size_t)(t0 + r) * HD + c0;
      _Float16* d = sv + r * 130 + c0;
#pragma unroll
      for (int j = 0; j < 16; ++j) d[j] = (_Float16)p[j]; }
    __syncthreads();
    v16h a[4], b[4];
#pragma unroll
    for (int i = 0; i < 4; ++i) a[i] = frag_aT(skp, 258, wm * 64 + i * 16, 0, lane);
#pragma unroll
    for (int j = 0; j < 4; ++j) b[j] = frag_b(sv, 130, 0, wn * 64 + j * 16, lane);
#pragma unroll
    for (int i = 0; i < 4; ++i)
#pragma unroll
      for (int j = 0; j < 4; ++j) acc[i][j] = wmma16(a[i], b[j], acc[i][j]);
    __syncthreads();
  }
  float* sc = Sc + (size_t)(h * NCHUNK + c) * FD * HD;
  const int msel = (lane >> 4) << 3, nsel = lane & 15;
#pragma unroll
  for (int i = 0; i < 4; ++i)
#pragma unroll
    for (int j = 0; j < 4; ++j)
#pragma unroll
      for (int r = 0; r < 8; ++r) {
        int m = wm * 64 + i * 16 + msel + r;
        int n = wn * 64 + j * 16 + nsel;
        sc[(size_t)m * HD + n] = acc[i][j][r];
      }
  // column sums of KP chunk
  float s = 0.f;
  for (int t = 0; t < CHUNK; ++t) s += kpb[(size_t)t * FD + tid];
  zc[(h * NCHUNK + c) * FD + tid] = s;
}

// Exclusive prefix over chunks of S and z
__global__ __launch_bounds__(256) void prefS_kernel(const float* __restrict__ Sc,
                                                    float* __restrict__ Sp) {
  int h = blockIdx.y;
  int e = blockIdx.x * 256 + threadIdx.x;  // 0..32767
  float acc = 0.f;
  for (int c = 0; c < NCHUNK; ++c) {
    size_t idx = (size_t)(h * NCHUNK + c) * (FD * HD) + e;
    Sp[idx] = acc;
    acc += Sc[idx];
  }
}
__global__ __launch_bounds__(256) void prefZ_kernel(const float* __restrict__ zc,
                                                    float* __restrict__ zp) {
  int h = blockIdx.x, f = threadIdx.x;
  float acc = 0.f;
  for (int c = 0; c < NCHUNK; ++c) {
    int idx = (h * NCHUNK + c) * FD + f;
    zp[idx] = acc;
    acc += zc[idx];
  }
}

// ---------------------------------------------------------------------------
// Pass C: per (head, chunk):
//   M = causal_mask(QP_c @ KP_c^T);  num = M@V_c + QP_c@S_prev
//   den = rowsum(M) + QP_c . z_prev; att = num/(den+eps) -> (n, d) layout
// ---------------------------------------------------------------------------
__global__ __launch_bounds__(256) void attn_kernel(const float* __restrict__ qp,
                                                   const float* __restrict__ kp,
                                                   const float* __restrict__ Vh,
                                                   const float* __restrict__ Sp,
                                                   const float* __restrict__ zp,
                                                   float* __restrict__ att) {
  __shared__ _Float16 Mm[128 * 130];
  __shared__ _Float16 bufA[128 * 34];
  __shared__ _Float16 bufB[128 * 34];  // also reused as [32][130] (4160 <= 4352)
  __shared__ float zs[FD];
  __shared__ float dens[CHUNK];
  const int h = blockIdx.x >> 3, c = blockIdx.x & 7;
  const int tid = threadIdx.x, lane = tid & 31, wave = tid >> 5;
  const int wm = wave >> 1, wn = wave & 1;
  const float* qpb = qp + (((size_t)h << 10) + c * CHUNK) * FD;
  const float* kpb = kp + (((size_t)h << 10) + c * CHUNK) * FD;
  const float* vb = Vh + (((size_t)h << 10) + c * CHUNK) * HD;
  const float* spb = Sp + (size_t)(h * NCHUNK + c) * FD * HD;
  zs[tid] = zp[(h * NCHUNK + c) * FD + tid];

  const int lrow = tid >> 1, lcol = (tid & 1) * 16;
  const int brow = tid >> 3, bcol = (tid & 7) * 16;
  const int msel = (lane >> 4) << 3, nsel = lane & 15;

  // ---- Phase 1: M = QP @ KP^T (K = FD = 256) ----
  v8f acc[2][4];
#pragma unroll
  for (int i = 0; i < 2; ++i)
#pragma unroll
    for (int j = 0; j < 4; ++j) acc[i][j] = (v8f){0, 0, 0, 0, 0, 0, 0, 0};
  for (int kk = 0; kk < 8; ++kk) {
    const float* ap = qpb + (size_t)lrow * FD + kk * 32 + lcol;
    const float* bp = kpb + (size_t)lrow * FD + kk * 32 + lcol;
    _Float16* da = bufA + lrow * 34 + lcol;
    _Float16* db = bufB + lrow * 34 + lcol;
#pragma unroll
    for (int j = 0; j < 16; ++j) { da[j] = (_Float16)ap[j]; db[j] = (_Float16)bp[j]; }
    __syncthreads();
    v16h a0 = frag_a(bufA, 34, wm * 32, 0, lane);
    v16h a1 = frag_a(bufA, 34, wm * 32 + 16, 0, lane);
#pragma unroll
    for (int j = 0; j < 4; ++j) {
      v16h b = frag_bt(bufB, 34, 0, wn * 64 + j * 16, lane);
      acc[0][j] = wmma16(a0, b, acc[0][j]);
      acc[1][j] = wmma16(a1, b, acc[1][j]);
    }
    __syncthreads();
  }
  // causal mask (keep j<=i within chunk), spill to LDS f16
#pragma unroll
  for (int i = 0; i < 2; ++i)
#pragma unroll
    for (int j = 0; j < 4; ++j)
#pragma unroll
      for (int r = 0; r < 8; ++r) {
        int ml = wm * 32 + i * 16 + msel + r;
        int nl = wn * 64 + j * 16 + nsel;
        float v = (nl <= ml) ? acc[i][j][r] : 0.f;
        Mm[ml * 130 + nl] = (_Float16)v;
      }
  __syncthreads();

  // ---- Phase 2: num = M @ V (K=128) + QP @ S_prev (K=256) ----
  v8f acc2[2][4];
#pragma unroll
  for (int i = 0; i < 2; ++i)
#pragma unroll
    for (int j = 0; j < 4; ++j) acc2[i][j] = (v8f){0, 0, 0, 0, 0, 0, 0, 0};
  for (int kk = 0; kk < 4; ++kk) {
    const float* bp = vb + (size_t)(kk * 32 + brow) * HD + bcol;
    _Float16* db = bufB + brow * 130 + bcol;
#pragma unroll
    for (int j = 0; j < 16; ++j) db[j] = (_Float16)bp[j];
    __syncthreads();
    v16h a0 = frag_a(Mm, 130, wm * 32, kk * 32, lane);
    v16h a1 = frag_a(Mm, 130, wm * 32 + 16, kk * 32, lane);
#pragma unroll
    for (int j = 0; j < 4; ++j) {
      v16h b = frag_b(bufB, 130, 0, wn * 64 + j * 16, lane);
      acc2[0][j] = wmma16(a0, b, acc2[0][j]);
      acc2[1][j] = wmma16(a1, b, acc2[1][j]);
    }
    __syncthreads();
  }
  for (int kk = 0; kk < 8; ++kk) {
    const float* ap = qpb + (size_t)lrow * FD + kk * 32 + lcol;
    _Float16* da = bufA + lrow * 34 + lcol;
#pragma unroll
    for (int j = 0; j < 16; ++j) da[j] = (_Float16)ap[j];
    const float* bp = spb + (size_t)(kk * 32 + brow) * HD + bcol;
    _Float16* db = bufB + brow * 130 + bcol;
#pragma unroll
    for (int j = 0; j < 16; ++j) db[j] = (_Float16)bp[j];
    __syncthreads();
    v16h a0 = frag_a(bufA, 34, wm * 32, 0, lane);
    v16h a1 = frag_a(bufA, 34, wm * 32 + 16, 0, lane);
#pragma unroll
    for (int j = 0; j < 4; ++j) {
      v16h b = frag_b(bufB, 130, 0, wn * 64 + j * 16, lane);
      acc2[0][j] = wmma16(a0, b, acc2[0][j]);
      acc2[1][j] = wmma16(a1, b, acc2[1][j]);
    }
    __syncthreads();
  }

  // ---- denominator ----
  if (tid < CHUNK) {
    float dv = 0.f;
    for (int j = 0; j < CHUNK; ++j) dv += (float)Mm[tid * 130 + j];
    const float* qr = qpb + (size_t)tid * FD;
    for (int f = 0; f < FD; ++f) dv += qr[f] * zs[f];
    dens[tid] = dv;
  }
  __syncthreads();

  // ---- epilogue: att[(c*128+m)][h*128+n] ----
#pragma unroll
  for (int i = 0; i < 2; ++i)
#pragma unroll
    for (int j = 0; j < 4; ++j)
#pragma unroll
      for (int r = 0; r < 8; ++r) {
        int ml = wm * 32 + i * 16 + msel + r;
        int nl = wn * 64 + j * 16 + nsel;
        float o = acc2[i][j][r] / (dens[ml] + 1e-8f);
        att[(size_t)(c * CHUNK + ml) * DMODEL + h * HD + nl] = o;
      }
}

// ---------------------------------------------------------------------------
// Host launcher
// ---------------------------------------------------------------------------
extern "C" void kernel_launch(void* const* d_in, const int* in_sizes, int n_in,
                              void* d_out, int out_size, void* d_ws, size_t ws_size,
                              hipStream_t stream) {
  const float* x    = (const float*)d_in[0];
  const float* spike= (const float*)d_in[1];
  const float* W1   = (const float*)d_in[2];
  const float* b1   = (const float*)d_in[3];
  const float* W2   = (const float*)d_in[4];
  const float* b2   = (const float*)d_in[5];
  const float* W3   = (const float*)d_in[6];
  const float* b3   = (const float*)d_in[7];
  const float* Wout = (const float*)d_in[8];
  const float* bout = (const float*)d_in[9];
  const float* g_sn = (const float*)d_in[10];
  const float* b_sn = (const float*)d_in[11];
  const float* Wf1  = (const float*)d_in[12];
  const float* bf1  = (const float*)d_in[13];
  const float* Wf2  = (const float*)d_in[14];
  const float* bf2  = (const float*)d_in[15];
  const float* g_ln = (const float*)d_in[16];
  const float* b_ln = (const float*)d_in[17];
  const float* sm   = (const float*)d_in[18];
  const float* g_sn2= (const float*)d_in[19];
  const float* b_sn2= (const float*)d_in[20];
  const float* proj = (const float*)d_in[21];

  float* ws    = (float*)d_ws;
  float* s_ln  = ws;            // 512
  float* h1    = ws + 1024;     // 1024
  float* h2    = ws + 2048;     // 1024
  float* pvec  = ws + 3072;     // 1024
  float* alpha = ws + 4096;     // 1024
  float* kpv   = ws + 5120;     // 1024
  float* kcv   = ws + 6144;     // 1024
  float* Af    = ws + 7168;     // 2048
  float* Bf    = ws + 9216;     // 2048
  float* zc    = ws + 11264;    // 16384
  float* zpf   = ws + 27648;    // 16384
  float* Qh    = ws + 65536;                    // 1M floats (h,n,hd)
  float* Vh    = Qh + (1u << 20);               // 1M
  float* qpbuf = Vh + (1u << 20);               // 2M (h,n,f)
  float* kpbuf = qpbuf + (2u << 20);            // 2M
  float* Sc    = kpbuf + (2u << 20);            // 2M (h,c,f,hd)
  float* Sp    = Sc + (2u << 20);               // 2M
  float* attb  = Sp + (2u << 20);               // 1M (n,d)
  (void)in_sizes; (void)n_in; (void)out_size; (void)ws_size;

  // spike path (rank-collapsed K construction)
  spike_ln_kernel<<<1, SD, 0, stream>>>(spike, g_sn, b_sn, s_ln);
  matvec_kernel<<<4, 256, 0, stream>>>(s_ln, Wf1, bf1, h1, SD, 1);
  matvec_kernel<<<4, 256, 0, stream>>>(h1, Wf2, bf2, h2, DMODEL, 0);
  stats_kernel<<<1, DMODEL, 0, stream>>>(h2, g_ln, b_ln, g_sn2, sm, pvec, alpha);
  kvec_kernel<<<4, 256, 0, stream>>>(pvec, b_sn2, W2, b2, kpv, kcv);
  featk_kernel<<<NHEAD, 256, 0, stream>>>(kpv, kcv, proj, Af, Bf);
  kpmat_kernel<<<NHEAD * N_TOK, 256, 0, stream>>>(alpha, Af, Bf, kpbuf);

  // Q and V dense GEMMs (WMMA), heads layout output
  gemm_xwT_kernel<<<dim3(8, 8), 256, 0, stream>>>(x, W1, b1, Qh, DMODEL, DMODEL, 1);
  gemm_xwT_kernel<<<dim3(8, 8), 256, 0, stream>>>(x, W3, b3, Vh, DMODEL, DMODEL, 1);

  // Q feature map (WMMA)
  featq_kernel<<<dim3(2, 8, NHEAD), 256, 0, stream>>>(Qh, proj, qpbuf);

  // chunked causal linear attention
  chunk_kv_kernel<<<NHEAD * NCHUNK, 256, 0, stream>>>(kpbuf, Vh, Sc, zc);
  prefS_kernel<<<dim3(128, NHEAD), 256, 0, stream>>>(Sc, Sp);
  prefZ_kernel<<<NHEAD, 256, 0, stream>>>(zc, zpf);
  attn_kernel<<<NHEAD * NCHUNK, 256, 0, stream>>>(qpbuf, kpbuf, Vh, Sp, zpf, attb);

  // output projection (WMMA)
  gemm_xwT_kernel<<<dim3(8, 8), 256, 0, stream>>>(attb, Wout, bout, (float*)d_out,
                                                  DMODEL, DMODEL, 0);
}
